// RPolicy_90769838833991
// MI455X (gfx1250) — matrix operand order repaired and checked by reading, not compiled, not run on hardware
//
#include <hip/hip_runtime.h>
#include <hip/hip_bf16.h>

// MI455X / gfx1250, wave32. D = A*B+C via v_wmma_f32_16x16x32_f16.

typedef _Float16 v16h __attribute__((ext_vector_type(16)));
typedef _Float16 v8h  __attribute__((ext_vector_type(8)));
typedef float    v8f  __attribute__((ext_vector_type(8)));

constexpr int kT = 65536;
constexpr int kD = 128;
constexpr int kH = 128;
constexpr int kF = 100;
constexpr int kA = 18;

__device__ __forceinline__ v8f wmma_f16(v16h a, v16h b, v8f c) {
  return __builtin_amdgcn_wmma_f32_16x16x32_f16(false, a, false, b, (short)0, c,
                                                false, false);
}

__device__ __forceinline__ float sigm(float x) { return 1.f / (1.f + expf(-x)); }

// ---------------------------------------------------------------------------
// Kernel 1: X = relu(obs@W1+b1); xg = X@Wg1_x + bg1 ; xc = X@Wc1_x + bc1
// (x-dependent GRU1 terms hoisted out of the sequential scan), stored f16.
// ---------------------------------------------------------------------------
__global__ void __launch_bounds__(256) k1_precompute(
    const float* __restrict__ obs, const float* __restrict__ W1,
    const float* __restrict__ b1,  const float* __restrict__ Wg1,
    const float* __restrict__ bg1, const float* __restrict__ Wc1,
    const float* __restrict__ bc1, _Float16* __restrict__ xgh,
    _Float16* __restrict__ xch) {
  __shared__ __attribute__((aligned(64))) _Float16 X16[16 * 128];
  const int lane = threadIdx.x & 31;
  const int wid  = threadIdx.x >> 5;   // 0..7
  const int l15  = lane & 15;
  const int o8   = (lane >> 4) << 3;   // A-frag K sub-block (ISA 16-bit A layout)
  const int kh16 = (lane >> 4) << 3;   // B-frag K sub-block (ISA 16-bit B layout)
  const int rb   = (lane >> 4) ? 8 : 0;

  // Loop-invariant B fragments held in VGPRs.
  v16h w1f[4], gaf[4], gbf[4], ccf[4];
  const int colH  = wid * 16 + l15;        // 0..127
  const int colGb = (wid + 8) * 16 + l15;  // 128..255
#pragma unroll
  for (int kc = 0; kc < 4; ++kc) {
#pragma unroll
    for (int j = 0; j < 16; ++j) {
      const int r = kc * 32 + kh16 + j + (j & 8);    // K row (x-part rows 0..127)
      w1f[kc][j] = (_Float16)W1[r * kH + colH];
      gaf[kc][j] = (_Float16)Wg1[r * 256 + colH];
      gbf[kc][j] = (_Float16)Wg1[r * 256 + colGb];
      ccf[kc][j] = (_Float16)Wc1[r * kH + colH];
    }
  }
  const float b1v  = b1[colH];
  const float bgav = bg1[colH];
  const float bgbv = bg1[colGb];
  const float bcv  = bc1[colH];

  for (int tile = blockIdx.x; tile < kT / 16; tile += gridDim.x) {
    const int t0 = tile * 16;
    {  // X tile -> LDS (f16)
      v8f acc;
#pragma unroll
      for (int r = 0; r < 8; ++r) acc[r] = b1v;
      const float* arow = obs + (size_t)(t0 + l15) * kD;
#pragma unroll
      for (int kc = 0; kc < 4; ++kc) {
        v16h a;
#pragma unroll
        for (int j = 0; j < 8; ++j) {
          a[j]     = (_Float16)arow[kc * 32 + o8 + j];
          a[8 + j] = (_Float16)arow[kc * 32 + 16 + o8 + j];
        }
        acc = wmma_f16(a, w1f[kc], acc);
      }
#pragma unroll
      for (int r = 0; r < 8; ++r) {
        float v = acc[r];
        X16[(rb + r) * 128 + colH] = (_Float16)(v > 0.f ? v : 0.f);
      }
    }
    __syncthreads();
    {  // xg (two tiles) and xc (one tile) from LDS X
      v8f ag, ab2, ac;
#pragma unroll
      for (int r = 0; r < 8; ++r) { ag[r] = bgav; ab2[r] = bgbv; ac[r] = bcv; }
#pragma unroll
      for (int kc = 0; kc < 4; ++kc) {
        const _Float16* p = &X16[l15 * 128 + kc * 32];
        v16h a;
#pragma unroll
        for (int j = 0; j < 8; ++j) { a[j] = p[o8 + j]; a[8 + j] = p[16 + o8 + j]; }
        ag  = wmma_f16(a, gaf[kc], ag);
        ab2 = wmma_f16(a, gbf[kc], ab2);
        ac  = wmma_f16(a, ccf[kc], ac);
      }
#pragma unroll
      for (int r = 0; r < 8; ++r) {
        const size_t tt = (size_t)(t0 + rb + r);
        xgh[tt * 256 + colH]  = (_Float16)ag[r];
        xgh[tt * 256 + colGb] = (_Float16)ab2[r];
        xch[tt * 128 + colH]  = (_Float16)ac[r];
      }
    }
    __syncthreads();
  }
}

// ---------------------------------------------------------------------------
// Kernel 2: persistent single-workgroup sequential GRU scan (latency-bound).
// 512 threads = 16 waves. All recurrent weights live in VGPRs as WMMA
// B-fragments. Hidden state (f16 copy for A-fragments, f32 master) in LDS.
// A-matrix = hidden vector replicated into all 16 rows, so every lane's
// c[0] holds one output column of the matvec after the chained WMMAs.
// ---------------------------------------------------------------------------
__global__ void __launch_bounds__(512) k2_scan(
    const float* __restrict__ Wg1, const float* __restrict__ Wc1,
    const float* __restrict__ Wg2, const float* __restrict__ bg2,
    const float* __restrict__ Wc2, const float* __restrict__ bc2,
    const _Float16* __restrict__ xgh, const _Float16* __restrict__ xch,
    _Float16* __restrict__ ysh, float* __restrict__ out_tail) {
  // LDS: [0:128)=h1, [128:256)=h2, [256:384)=r1*h1, [384:512)=r2*h2 (f16)
  __shared__ __attribute__((aligned(64))) _Float16 Hh[512];
  __shared__ float h1f[128], h2f[128], u1f[128], u2f[128];

  const int lane = threadIdx.x & 31;
  const int wid  = threadIdx.x >> 5;  // 0..15
  const int l15  = lane & 15;
  const int o8   = (lane >> 4) << 3;
  const int kh16 = (lane >> 4) << 3;

  for (int i = threadIdx.x; i < 512; i += 512) Hh[i] = (_Float16)0.f;
  if (threadIdx.x < 128) { h1f[threadIdx.x] = 0.f; h2f[threadIdx.x] = 0.f; }

  const int gcol = wid * 16 + l15;          // gate column 0..255
  const int ccol = (wid & 7) * 16 + l15;    // candidate column 0..127

  // --- resident weight fragments (~160 VGPRs/wave) ---
  v16h g1f[4];  // Wg1 h-part (rows 128..255), col gcol
#pragma unroll
  for (int kc = 0; kc < 4; ++kc)
#pragma unroll
    for (int j = 0; j < 16; ++j)
      g1f[kc][j] = (_Float16)Wg1[(128 + kc * 32 + kh16 + j + (j & 8)) * 256 + gcol];
  v16h g2f[8];  // Wg2 (rows 0..255 over [h1;h2]), col gcol
#pragma unroll
  for (int kc = 0; kc < 8; ++kc)
#pragma unroll
    for (int j = 0; j < 16; ++j)
      g2f[kc][j] = (_Float16)Wg2[(kc * 32 + kh16 + j + (j & 8)) * 256 + gcol];
  v16h cf[8];   // waves 0-7: Wc1 h-part (4 chunks); waves 8-15: Wc2 (8 chunks)
  if (wid < 8) {
#pragma unroll
    for (int kc = 0; kc < 4; ++kc)
#pragma unroll
      for (int j = 0; j < 16; ++j)
        cf[kc][j] = (_Float16)Wc1[(128 + kc * 32 + kh16 + j + (j & 8)) * 128 + ccol];
  } else {
#pragma unroll
    for (int kc = 0; kc < 8; ++kc)
#pragma unroll
      for (int j = 0; j < 16; ++j)
        cf[kc][j] = (_Float16)Wc2[(kc * 32 + kh16 + j + (j & 8)) * 128 + ccol];
  }
  const float bg2v = bg2[gcol];
  const float bc2v = bc2[ccol];
  __syncthreads();

  // double-buffered prefetch of the precomputed x-terms
  float xg_cur = (float)xgh[gcol];
  float xc_cur = (wid < 8) ? (float)xch[ccol] : 0.f;

  for (int t = 0; t < kT; ++t) {
    const int tn = (t + 1 < kT) ? t + 1 : t;
    const _Float16 xg_nx = xgh[(size_t)tn * 256 + gcol];
    const _Float16 xc_nx = (wid < 8) ? xch[(size_t)tn * 128 + ccol] : (_Float16)0.f;

    // ---- GRU1 gates: z = xg[t] + h1 @ Wg1_h ----
    {
      v8f acc = {};
#pragma unroll
      for (int kc = 0; kc < 4; ++kc) {
        const _Float16* p = &Hh[kc * 32];
        v16h a;
#pragma unroll
        for (int j = 0; j < 8; ++j) { a[j] = p[o8 + j]; a[8 + j] = p[16 + o8 + j]; }
        acc = wmma_f16(a, g1f[kc], acc);
      }
      const float s = sigm(acc[0] + xg_cur);
      if (wid < 8) Hh[256 + gcol] = (_Float16)(s * h1f[gcol]);   // r1*h1
      else         u1f[gcol - 128] = s;                           // u1
    }
    __syncthreads();
    // ---- GRU1 candidate + h1 update (waves 0..7) ----
    if (wid < 8) {
      v8f acc = {};
#pragma unroll
      for (int kc = 0; kc < 4; ++kc) {
        const _Float16* p = &Hh[256 + kc * 32];
        v16h a;
#pragma unroll
        for (int j = 0; j < 8; ++j) { a[j] = p[o8 + j]; a[8 + j] = p[16 + o8 + j]; }
        acc = wmma_f16(a, cf[kc], acc);
      }
      const float cv = tanhf(acc[0] + xc_cur);
      const float u  = u1f[ccol];
      const float hn = u * h1f[ccol] + (1.f - u) * cv;
      h1f[ccol] = hn;
      Hh[ccol]  = (_Float16)hn;
    }
    __syncthreads();
    // ---- GRU2 gates: z = bg2 + [h1;h2] @ Wg2 ----
    {
      v8f acc = {};
#pragma unroll
      for (int kc = 0; kc < 8; ++kc) {
        const _Float16* p = &Hh[kc * 32];
        v16h a;
#pragma unroll
        for (int j = 0; j < 8; ++j) { a[j] = p[o8 + j]; a[8 + j] = p[16 + o8 + j]; }
        acc = wmma_f16(a, g2f[kc], acc);
      }
      const float s = sigm(acc[0] + bg2v);
      if (wid < 8) Hh[384 + gcol] = (_Float16)(s * h2f[gcol]);   // r2*h2
      else         u2f[gcol - 128] = s;                           // u2
    }
    __syncthreads();
    // ---- GRU2 candidate + h2 update + ys[t] (waves 8..15) ----
    if (wid >= 8) {
      v8f acc = {};
#pragma unroll
      for (int kc = 0; kc < 8; ++kc) {
        const _Float16* p = (kc < 4) ? &Hh[kc * 32] : &Hh[384 + (kc - 4) * 32];
        v16h a;
#pragma unroll
        for (int j = 0; j < 8; ++j) { a[j] = p[o8 + j]; a[8 + j] = p[16 + o8 + j]; }
        acc = wmma_f16(a, cf[kc], acc);
      }
      const float cv = tanhf(acc[0] + bc2v);
      const float u  = u2f[ccol];
      const float hn = u * h2f[ccol] + (1.f - u) * cv;
      h2f[ccol]      = hn;
      Hh[128 + ccol] = (_Float16)hn;
      if (lane < 16) ysh[(size_t)t * 128 + ccol] = (_Float16)hn;
    }
    __syncthreads();
    xg_cur = (float)xg_nx;
    xc_cur = (wid < 8) ? (float)xc_nx : 0.f;
  }
  if (threadIdx.x < 128) {
    out_tail[threadIdx.x]       = h1f[threadIdx.x];
    out_tail[128 + threadIdx.x] = h2f[threadIdx.x];
  }
}

// ---------------------------------------------------------------------------
// Kernel 3: heads. pi=relu(ys@W2+b2), v=relu(ys@W3+b3) (F padded 100->112/128
// with zeros), logits=pi@Wp+bp, value=v@Wv+bv.
// ---------------------------------------------------------------------------
__global__ void __launch_bounds__(256) k3_heads(
    const _Float16* __restrict__ ysh,
    const float* __restrict__ W2, const float* __restrict__ b2,
    const float* __restrict__ W3, const float* __restrict__ b3,
    const float* __restrict__ Wp, const float* __restrict__ bp,
    const float* __restrict__ Wv, const float* __restrict__ bv,
    float* __restrict__ logits, float* __restrict__ value) {
  __shared__ __attribute__((aligned(64))) _Float16 P16[16 * 128];
  __shared__ __attribute__((aligned(64))) _Float16 V16[16 * 128];
  const int lane = threadIdx.x & 31;
  const int wid  = threadIdx.x >> 5;  // 0..7
  const int l15  = lane & 15;
  const int o8   = (lane >> 4) << 3;
  const int kh16 = (lane >> 4) << 3;
  const int rb   = (lane >> 4) ? 8 : 0;

  for (int i = threadIdx.x; i < 16 * 128; i += 256) {
    P16[i] = (_Float16)0.f;
    V16[i] = (_Float16)0.f;
  }

  v16h w2f[4] = {}, w3f[4] = {};
  float b2v = 0.f, b3v = 0.f;
  if (wid < 7) {  // waves 0..6: pi tile wid and v tile wid (cols wid*16..)
    const int col = wid * 16 + l15;
    const bool ok = col < kF;
#pragma unroll
    for (int kc = 0; kc < 4; ++kc)
#pragma unroll
      for (int j = 0; j < 16; ++j) {
        const int r = kc * 32 + kh16 + j + (j & 8);
        w2f[kc][j] = ok ? (_Float16)W2[r * kF + col] : (_Float16)0.f;
        w3f[kc][j] = ok ? (_Float16)W3[r * kF + col] : (_Float16)0.f;
      }
    b2v = ok ? b2[col] : 0.f;
    b3v = ok ? b3[col] : 0.f;
  }
  v16h pf[4] = {};
  float bpv = 0.f;
  if (wid < 2) {  // logits tiles (cols 0-15, 16-31; valid < 18)
    const int col = wid * 16 + l15;
    const bool ok = col < kA;
#pragma unroll
    for (int kc = 0; kc < 4; ++kc)
#pragma unroll
      for (int j = 0; j < 16; ++j) {
        const int r = kc * 32 + kh16 + j + (j & 8);  // K over padded F (rows>=100 -> 0)
        pf[kc][j] = (ok && r < kF) ? (_Float16)Wp[r * kA + col] : (_Float16)0.f;
      }
    bpv = ok ? bp[col] : 0.f;
  } else if (wid == 2) {  // value (col 0 only)
    const bool ok = (l15 == 0);
#pragma unroll
    for (int kc = 0; kc < 4; ++kc)
#pragma unroll
      for (int j = 0; j < 16; ++j) {
        const int r = kc * 32 + kh16 + j + (j & 8);
        pf[kc][j] = (ok && r < kF) ? (_Float16)Wv[r] : (_Float16)0.f;
      }
    bpv = bv[0];
  }
  __syncthreads();

  for (int tile = blockIdx.x; tile < kT / 16; tile += gridDim.x) {
    const int t0 = tile * 16;
    if (wid < 7) {
      v8f ap, av;
#pragma unroll
      for (int r = 0; r < 8; ++r) { ap[r] = b2v; av[r] = b3v; }
      const _Float16* arow = ysh + (size_t)(t0 + l15) * 128;
#pragma unroll
      for (int kc = 0; kc < 4; ++kc) {
        const _Float16* p = arow + kc * 32;
        v16h a;
#pragma unroll
        for (int j = 0; j < 8; ++j) { a[j] = p[o8 + j]; a[8 + j] = p[16 + o8 + j]; }
        ap = wmma_f16(a, w2f[kc], ap);
        av = wmma_f16(a, w3f[kc], av);
      }
      const int col = wid * 16 + l15;
#pragma unroll
      for (int r = 0; r < 8; ++r) {
        float pv = ap[r], vv = av[r];
        P16[(rb + r) * 128 + col] = (_Float16)(pv > 0.f ? pv : 0.f);
        V16[(rb + r) * 128 + col] = (_Float16)(vv > 0.f ? vv : 0.f);
      }
    }
    __syncthreads();
    if (wid < 2) {
      v8f acc;
#pragma unroll
      for (int r = 0; r < 8; ++r) acc[r] = bpv;
#pragma unroll
      for (int kc = 0; kc < 4; ++kc) {
        const _Float16* p = &P16[l15 * 128 + kc * 32];
        v16h a;
#pragma unroll
        for (int j = 0; j < 8; ++j) { a[j] = p[o8 + j]; a[8 + j] = p[16 + o8 + j]; }
        acc = wmma_f16(a, pf[kc], acc);
      }
      const int col = wid * 16 + l15;
      if (col < kA) {
#pragma unroll
        for (int r = 0; r < 8; ++r)
          logits[(size_t)(t0 + rb + r) * kA + col] = acc[r];
      }
    } else if (wid == 2) {
      v8f acc;
#pragma unroll
      for (int r = 0; r < 8; ++r) acc[r] = bpv;
#pragma unroll
      for (int kc = 0; kc < 4; ++kc) {
        const _Float16* p = &V16[l15 * 128 + kc * 32];
        v16h a;
#pragma unroll
        for (int j = 0; j < 8; ++j) { a[j] = p[o8 + j]; a[8 + j] = p[16 + o8 + j]; }
        acc = wmma_f16(a, pf[kc], acc);
      }
      if (l15 == 0) {
#pragma unroll
        for (int r = 0; r < 8; ++r) value[t0 + rb + r] = acc[r];
      }
    }
    __syncthreads();
  }
}

// ---------------------------------------------------------------------------
extern "C" void kernel_launch(void* const* d_in, const int* in_sizes, int n_in,
                              void* d_out, int out_size, void* d_ws,
                              size_t ws_size, hipStream_t stream) {
  (void)in_sizes; (void)n_in; (void)out_size; (void)ws_size;
  const float* obs = (const float*)d_in[0];
  const float* W1  = (const float*)d_in[1];
  const float* b1  = (const float*)d_in[2];
  const float* Wg1 = (const float*)d_in[3];
  const float* bg1 = (const float*)d_in[4];
  const float* Wc1 = (const float*)d_in[5];
  const float* bc1 = (const float*)d_in[6];
  const float* Wg2 = (const float*)d_in[7];
  const float* bg2 = (const float*)d_in[8];
  const float* Wc2 = (const float*)d_in[9];
  const float* bc2 = (const float*)d_in[10];
  const float* W2  = (const float*)d_in[11];
  const float* b2  = (const float*)d_in[12];
  const float* W3  = (const float*)d_in[13];
  const float* b3  = (const float*)d_in[14];
  const float* Wp  = (const float*)d_in[15];
  const float* bp  = (const float*)d_in[16];
  const float* Wv  = (const float*)d_in[17];
  const float* bv  = (const float*)d_in[18];

  float* out    = (float*)d_out;
  float* logits = out;                       // [T,18]
  float* value  = out + (size_t)kT * kA;     // [T,1]
  float* tail   = out + (size_t)kT * (kA+1); // h1[128], h2[128]

  char* ws = (char*)d_ws;                                   // 64 MB total (f16)
  _Float16* xgh = (_Float16*)ws;                            // [T,256]
  _Float16* xch = (_Float16*)(ws + (size_t)kT * 256 * 2);   // [T,128]
  _Float16* ysh = (_Float16*)(ws + (size_t)kT * 256 * 2 + (size_t)kT * 128 * 2);

  k1_precompute<<<dim3(1024), dim3(256), 0, stream>>>(obs, W1, b1, Wg1, bg1,
                                                      Wc1, bc1, xgh, xch);
  k2_scan<<<dim3(1), dim3(512), 0, stream>>>(Wg1, Wc1, Wg2, bg2, Wc2, bc2,
                                             xgh, xch, ysh, tail);
  k3_heads<<<dim3(1024), dim3(256), 0, stream>>>(ysh, W2, b2, W3, b3, Wp, bp,
                                                 Wv, bv, logits, value);
}
